// ONINorm_38826504356590
// MI455X (gfx1250) — matrix operand order repaired and checked by reading, not compiled, not run on hardware
//
#include <hip/hip_runtime.h>
#include <math.h>

// ---------------------------------------------------------------------------
// ONI (group-whitening / Newton-Schulz orthogonalization) for MI455X (gfx1250)
// All GEMMs: bf16 inputs (converted in LDS loaders), fp32 accumulation via
// v_wmma_f32_16x16x32_bf16 (wave32, 16x16x32 tiles).
// LDS layouts: A tile m-major (pairs contiguous -> ds_load_2addr_b32),
//              B tile n-major K-contiguous (32B runs -> ds_load_b128 pairs).
// 32-bit offsets -> SGPR-base + v32 scale_offset addressing.
// Tile loads staged through registers BEFORE the barrier: batched load
// clauses overlap the previous K-step's WMMAs; only cvt+ds_store sit
// between the two barriers.
// ---------------------------------------------------------------------------

typedef __attribute__((ext_vector_type(16))) __bf16 v16bf;
typedef __attribute__((ext_vector_type(8)))  float  v8f;

#define G4        4       // NORM_GROUPS
#define DD        512u    // d = C / g
#define NR        16384u  // samples per group
#define SPLITK    4       // gram split-K partials (deterministic reduce)
#define ROWCHUNKS 64      // colsum stage-1 chunks of 256 rows
#define BM        128u
#define BN        128u
#define BK        32u
#define LDTA      (BK + 2u)  // A tile stride (bf16): 68B rows, 4B-aligned pairs
#define LDTB      (BK + 8u)  // B tile stride (bf16): 80B rows, 16B-aligned
#define EPSV      1e-5f

// --------------------------- fragment gathers ------------------------------
// A 16x32 bf16 fragment: lane L holds row m=L&15; lanes 0-15 get
// K=0..15; lanes 16-31 get K=16..31 (matches gather_bt).
__device__ inline v16bf gather_a(const __bf16* ls, unsigned mo, unsigned lane) {
  const unsigned half = lane >> 4, l15 = lane & 15;
  const __bf16* row = ls + (mo + l15) * LDTA + half * 16;
  v16bf a;
#pragma unroll
  for (int t = 0; t < 8; ++t) { a[t] = row[t]; a[t + 8] = row[8 + t]; }
  return a;
}

// B 32x16 bf16 fragment: lane L holds col n=L&15; lanes 0-15 hold K=0..15,
// lanes 16-31 hold K=16..31; VGPR v holds K=2v (lo) and K=2v+1 (hi).
// With the n-major K-contiguous LDS layout this is one 32-byte contiguous run.
__device__ inline v16bf gather_bt(const __bf16* ls, unsigned no, unsigned lane) {
  const unsigned half = lane >> 4, l15 = lane & 15;
  const __bf16* row = ls + (no + l15) * LDTB + half * 16;
  v16bf b;
#pragma unroll
  for (int t = 0; t < 16; ++t) b[t] = row[t];
  return b;
}

// 8 WMMAs per wave per K-step: 2 A-frags x 4 B-frags (32x64 wave macro-tile)
__device__ inline void wave_mma(const __bf16* lsA, const __bf16* lsBt,
                                unsigned wm, unsigned wn, unsigned lane,
                                v8f acc[2][4]) {
  v16bf bf[4];
#pragma unroll
  for (int j = 0; j < 4; ++j) bf[j] = gather_bt(lsBt, wn + 16 * j, lane);
#pragma unroll
  for (int i = 0; i < 2; ++i) {
    v16bf af = gather_a(lsA, wm + 16 * i, lane);
#pragma unroll
    for (int j = 0; j < 4; ++j)
      acc[i][j] = __builtin_amdgcn_wmma_f32_16x16x32_bf16(
          false, af, false, bf[j], (short)0, acc[i][j], false, false);
  }
}

// --------------------------- column sums -----------------------------------
__global__ __launch_bounds__(256) void colsum_part_kernel(
    const float* __restrict__ Z, float* __restrict__ part) {
  const unsigned chunk = blockIdx.x, g = blockIdx.y, t = threadIdx.x;
  const float* Zg = Z + (size_t)g * NR * DD;
  float a0 = 0.f, a1 = 0.f;
  const unsigned r0 = chunk * 256u;
  for (unsigned r = 0; r < 256u; ++r) {
    const unsigned ro = (r0 + r) * DD;
    a0 += Zg[ro + t];
    a1 += Zg[ro + t + 256u];
  }
  float* p = part + ((size_t)g * ROWCHUNKS + chunk) * DD;
  p[t] = a0;
  p[t + 256u] = a1;
}

__global__ __launch_bounds__(256) void colsum_final_kernel(
    const float* __restrict__ part, float* __restrict__ s) {
  const unsigned c = blockIdx.x * 256u + threadIdx.x;  // 0..2047
  const unsigned g = c >> 9, k = c & (DD - 1u);
  float a = 0.f;
  for (unsigned ch = 0; ch < ROWCHUNKS; ++ch)
    a += part[(g * ROWCHUNKS + ch) * DD + k];
  s[c] = a;
}

// --------------------------- Gram: G = Z^T Z -------------------------------
// grid: (16 tiles, SPLITK, G4). Each block owns a disjoint 128x128 tile of
// one split-K partial buffer -> no atomics, deterministic.
__global__ __launch_bounds__(256) void gram_kernel(
    const float* __restrict__ Z, float* __restrict__ Gpart) {
  const unsigned tile = blockIdx.x, p = blockIdx.y, g = blockIdx.z;
  const unsigned ti = tile & 3u, tj = tile >> 2;
  const float* Zg = Z + (size_t)g * NR * DD;
  float* Dst = Gpart + ((size_t)p * G4 + g) * DD * DD;

  __shared__ __align__(16) __bf16 lsA[BM * LDTA];
  __shared__ __align__(16) __bf16 lsBt[BN * LDTB];

  const unsigned tid = threadIdx.x, lane = tid & 31u, wave = tid >> 5;
  const unsigned wm = (wave & 3u) * 32u, wn = (wave >> 2) * 64u;

  v8f acc[2][4];
  const v8f vz = {0.f, 0.f, 0.f, 0.f, 0.f, 0.f, 0.f, 0.f};
#pragma unroll
  for (int i = 0; i < 2; ++i)
#pragma unroll
    for (int j = 0; j < 4; ++j) acc[i][j] = vz;

  const unsigned n0 = p * (NR / SPLITK);
  const unsigned nend = n0 + NR / SPLITK;
  for (unsigned nn = n0; nn < nend; nn += BK) {
    // Stage tiles into registers (batched global loads, overlaps prior WMMAs)
    float ta[16], tb[16];
#pragma unroll
    for (unsigned u = 0; u < 16u; ++u) {
      unsigned e = u * 256u + tid, k = e >> 7, x = e & 127u;
      unsigned ro = (nn + k) * DD;
      ta[u] = Zg[ro + ti * BM + x];  // -> A'[x][k] (transposed)
      tb[u] = Zg[ro + tj * BN + x];  // -> B'[x][k]
    }
    __syncthreads();
#pragma unroll
    for (unsigned u = 0; u < 16u; ++u) {
      unsigned e = u * 256u + tid, k = e >> 7, x = e & 127u;
      lsA[x * LDTA + k] = (__bf16)ta[u];
      lsBt[x * LDTB + k] = (__bf16)tb[u];
    }
    __syncthreads();
    wave_mma(lsA, lsBt, wm, wn, lane, acc);
  }

  const unsigned half = lane >> 4, l15 = lane & 15u;
#pragma unroll
  for (unsigned i = 0; i < 2; ++i)
#pragma unroll
    for (unsigned j = 0; j < 4; ++j) {
      const unsigned col = tj * BN + wn + 16u * j + l15;
#pragma unroll
      for (unsigned v = 0; v < 8; ++v) {
        const unsigned row = ti * BM + wm + 16u * i + v + 8u * half;
        Dst[row * DD + col] = acc[i][j][v];
      }
    }
}

// ---- S = sum(Gpart) - s s^T / N + eps I; Frobenius-normalize; rn = ||S||^-1/2
__global__ __launch_bounds__(1024) void finishS_kernel(
    const float* __restrict__ Gpart, const float* __restrict__ s,
    float* __restrict__ S, float* __restrict__ rn) {
  const unsigned g = blockIdx.x, tid = threadIdx.x;
  const float* sg = s + g * DD;
  float* Sg = S + (size_t)g * DD * DD;
  const float* Gg = Gpart + (size_t)g * DD * DD;
  const unsigned pstride = G4 * DD * DD;  // 4M elements < 2^31
  const float invN = 1.0f / (float)NR;
  float sq = 0.f;
  for (unsigned idx = tid; idx < DD * DD; idx += 1024u) {
    const unsigned i = idx >> 9, j = idx & (DD - 1u);
    float v = Gg[idx] + Gg[idx + pstride] + Gg[idx + 2u * pstride] +
              Gg[idx + 3u * pstride];
    v -= sg[i] * sg[j] * invN;
    if (i == j) v += EPSV;
    Sg[idx] = v;
    sq += v * v;
  }
  __shared__ float red[1024];
  red[tid] = sq;
  __syncthreads();
  for (unsigned st = 512; st > 0; st >>= 1) {
    if (tid < st) red[tid] += red[tid + st];
    __syncthreads();
  }
  const float norm = sqrtf(red[0]);
  const float inv = 1.0f / norm;
  if (tid == 0) rn[g] = rsqrtf(norm);
  for (unsigned idx = tid; idx < DD * DD; idx += 1024u) Sg[idx] *= inv;
}

// --------------------------- B0 = I ----------------------------------------
__global__ __launch_bounds__(256) void eye_kernel(float* __restrict__ B0) {
  const unsigned idx = blockIdx.x * 256u + threadIdx.x;
  if (idx < G4 * DD * DD) {
    const unsigned e = idx & (DD * DD - 1u);
    B0[idx] = ((e >> 9) == (e & (DD - 1u))) ? 1.0f : 0.0f;
  }
}

// --------- batched 512x512x512 GEMM: D = alpha*(A@B) + beta*Cadd -----------
__global__ __launch_bounds__(256) void mm512_kernel(
    const float* __restrict__ A, const float* __restrict__ Bm,
    const float* __restrict__ Cadd, float* __restrict__ Dm,
    float alpha, float beta) {
  const unsigned tn = blockIdx.x, tm = blockIdx.y, g = blockIdx.z;
  const size_t gb = (size_t)g * DD * DD;
  A += gb; Bm += gb; Dm += gb;
  if (Cadd) Cadd += gb;

  __shared__ __align__(16) __bf16 lsA[BM * LDTA];
  __shared__ __align__(16) __bf16 lsBt[BN * LDTB];

  const unsigned tid = threadIdx.x, lane = tid & 31u, wave = tid >> 5;
  const unsigned wm = (wave & 3u) * 32u, wn = (wave >> 2) * 64u;

  v8f acc[2][4];
  const v8f vz = {0.f, 0.f, 0.f, 0.f, 0.f, 0.f, 0.f, 0.f};
#pragma unroll
  for (int i = 0; i < 2; ++i)
#pragma unroll
    for (int j = 0; j < 4; ++j) acc[i][j] = vz;

  const unsigned abase = tm * BM * DD;
  const unsigned bbase = tn * BN;

  for (unsigned kk = 0; kk < DD; kk += BK) {
    float ta[16], tb[16];
#pragma unroll
    for (unsigned u = 0; u < 16u; ++u) {
      unsigned e = u * 256u + tid, r = e >> 5, c = e & 31u;
      ta[u] = A[abase + r * DD + kk + c];
    }
#pragma unroll
    for (unsigned u = 0; u < 16u; ++u) {
      unsigned e = u * 256u + tid, k = e >> 7, n = e & 127u;
      tb[u] = Bm[bbase + (kk + k) * DD + n];
    }
    __syncthreads();
#pragma unroll
    for (unsigned u = 0; u < 16u; ++u) {
      unsigned e = u * 256u + tid, r = e >> 5, c = e & 31u;
      lsA[r * LDTA + c] = (__bf16)ta[u];
    }
#pragma unroll
    for (unsigned u = 0; u < 16u; ++u) {
      unsigned e = u * 256u + tid, k = e >> 7, n = e & 127u;
      lsBt[n * LDTB + k] = (__bf16)tb[u];
    }
    __syncthreads();
    wave_mma(lsA, lsBt, wm, wn, lane, acc);
  }

  const unsigned half = lane >> 4, l15 = lane & 15u;
#pragma unroll
  for (unsigned i = 0; i < 2; ++i)
#pragma unroll
    for (unsigned j = 0; j < 4; ++j) {
      const unsigned col = tn * BN + wn + 16u * j + l15;
#pragma unroll
      for (unsigned v = 0; v < 8; ++v) {
        const unsigned row = tm * BM + wm + 16u * i + v + 8u * half;
        const unsigned off = row * DD + col;
        const float cv = Cadd ? Cadd[off] : 0.f;
        Dm[off] = alpha * acc[i][j][v] + beta * cv;
      }
    }
}

// ------------- Out = (Z - mu) @ B^T * rsqrt(||S||_F) -----------------------
__global__ __launch_bounds__(256) void out_kernel(
    const float* __restrict__ Z, const float* __restrict__ Bm,
    const float* __restrict__ s, const float* __restrict__ rn,
    float* __restrict__ O) {
  const unsigned tn = blockIdx.x, tm = blockIdx.y, g = blockIdx.z;
  const float* Zg = Z + (size_t)g * NR * DD;
  float* Og = O + (size_t)g * NR * DD;
  const float* Bg = Bm + (size_t)g * DD * DD;
  const float* sg = s + g * DD;
  const float invN = 1.0f / (float)NR;
  const float scale = rn[g];

  __shared__ __align__(16) __bf16 lsA[BM * LDTA];
  __shared__ __align__(16) __bf16 lsBt[BN * LDTB];

  const unsigned tid = threadIdx.x, lane = tid & 31u, wave = tid >> 5;
  const unsigned wm = (wave & 3u) * 32u, wn = (wave >> 2) * 64u;

  v8f acc[2][4];
  const v8f vz = {0.f, 0.f, 0.f, 0.f, 0.f, 0.f, 0.f, 0.f};
#pragma unroll
  for (int i = 0; i < 2; ++i)
#pragma unroll
    for (int j = 0; j < 4; ++j) acc[i][j] = vz;

  for (unsigned kk = 0; kk < DD; kk += BK) {
    float ta[16], tmu[16], tb[16];
#pragma unroll
    for (unsigned u = 0; u < 16u; ++u) {
      unsigned e = u * 256u + tid, r = e >> 5, c = e & 31u;
      ta[u] = Zg[(tm * BM + r) * DD + kk + c];
      tmu[u] = sg[kk + c];
    }
#pragma unroll
    for (unsigned u = 0; u < 16u; ++u) {
      unsigned e = u * 256u + tid, n = e >> 5, c = e & 31u;
      tb[u] = Bg[(tn * BN + n) * DD + kk + c];
    }
    __syncthreads();
#pragma unroll
    for (unsigned u = 0; u < 16u; ++u) {
      unsigned e = u * 256u + tid, r = e >> 5, c = e & 31u;
      lsA[r * LDTA + c] = (__bf16)(ta[u] - tmu[u] * invN);
    }
#pragma unroll
    for (unsigned u = 0; u < 16u; ++u) {
      unsigned e = u * 256u + tid, n = e >> 5, c = e & 31u;
      lsBt[n * LDTB + c] = (__bf16)tb[u];
    }
    __syncthreads();
    wave_mma(lsA, lsBt, wm, wn, lane, acc);
  }

  const unsigned half = lane >> 4, l15 = lane & 15u;
#pragma unroll
  for (unsigned i = 0; i < 2; ++i)
#pragma unroll
    for (unsigned j = 0; j < 4; ++j) {
      const unsigned col = tn * BN + wn + 16u * j + l15;
#pragma unroll
      for (unsigned v = 0; v < 8; ++v) {
        const unsigned row = tm * BM + wm + 16u * i + v + 8u * half;
        Og[row * DD + col] = scale * acc[i][j][v];
      }
    }
}

// --------------------------- host-side launcher ----------------------------
extern "C" void kernel_launch(void* const* d_in, const int* in_sizes, int n_in,
                              void* d_out, int out_size, void* d_ws,
                              size_t ws_size, hipStream_t stream) {
  (void)in_sizes; (void)n_in; (void)out_size; (void)ws_size;
  const float* Z = (const float*)d_in[0];
  float* O = (float*)d_out;

  // workspace layout (floats); everything fully rewritten each call
  float* w = (float*)d_ws;
  float* colpart = w;  w += (size_t)G4 * ROWCHUNKS * DD;    // 128K floats
  float* s = w;        w += (size_t)G4 * DD;                // 2K
  float* Gpart = w;    w += (size_t)SPLITK * G4 * DD * DD;  // 4M
  float* S = w;        w += (size_t)G4 * DD * DD;           // 1M
  float* B0 = w;       w += (size_t)G4 * DD * DD;           // 1M
  float* B1 = w;       w += (size_t)G4 * DD * DD;           // 1M
  float* T1 = w;       w += (size_t)G4 * DD * DD;           // 1M
  float* T2 = w;       w += (size_t)G4 * DD * DD;           // 1M
  float* rn = w;       w += G4;

  // 1) column sums (two-stage, deterministic)
  colsum_part_kernel<<<dim3(ROWCHUNKS, G4), 256, 0, stream>>>(Z, colpart);
  colsum_final_kernel<<<dim3((G4 * DD) / 256u), 256, 0, stream>>>(colpart, s);

  // 2) Gram Z^T Z via bf16 WMMA, split-K partials
  gram_kernel<<<dim3(16, SPLITK, G4), 256, 0, stream>>>(Z, Gpart);

  // 3) S = Gram - s s^T/N + eps I, Frobenius-normalized; rn = ||S||^-1/2
  finishS_kernel<<<dim3(G4), 1024, 0, stream>>>(Gpart, s, S, rn);

  // 4) Newton-Schulz: B <- 1.5 B - 0.5 (B B B) S, 5 iterations
  eye_kernel<<<dim3((G4 * DD * DD) / 256u), 256, 0, stream>>>(B0);
  float* Bc = B0;
  float* Bn = B1;
  for (int t = 0; t < 5; ++t) {
    mm512_kernel<<<dim3(4, 4, G4), 256, 0, stream>>>(Bc, Bc, nullptr, T1, 1.0f, 0.0f);
    mm512_kernel<<<dim3(4, 4, G4), 256, 0, stream>>>(T1, Bc, nullptr, T2, 1.0f, 0.0f);
    mm512_kernel<<<dim3(4, 4, G4), 256, 0, stream>>>(T2, S, Bc, Bn, -0.5f, 1.5f);
    float* tmp = Bc; Bc = Bn; Bn = tmp;
  }

  // 5) Out = Zc @ B^T * rsqrt(norm), back in the reference's flat layout
  out_kernel<<<dim3(4, NR / BM, G4), 256, 0, stream>>>(Z, Bc, s, rn, O);
}
